// TreeClassificationModel_31980326486214
// MI455X (gfx1250) — matrix operand order repaired and checked by reading, not compiled, not run on hardware
//
#include <hip/hip_runtime.h>
#include <hip/hip_bf16.h>

// ---------------------------------------------------------------------------
// Types for CDNA5 WMMA (wave32): v_wmma_f32_16x16x32_bf16
// ---------------------------------------------------------------------------
typedef __bf16 v16bf __attribute__((ext_vector_type(16)));
typedef float  v8f   __attribute__((ext_vector_type(8)));

union FragU { unsigned int u[8]; v16bf v; };

__device__ __forceinline__ unsigned int f2bf(float f) {
  // round-to-nearest-even f32 -> bf16 (result in low 16 bits)
  unsigned int u = __float_as_uint(f);
  unsigned int r = u + 0x7FFFu + ((u >> 16) & 1u);
  return r >> 16;
}
__device__ __forceinline__ unsigned int pack2bf(float lo, float hi) {
#if defined(__has_builtin)
#if __has_builtin(__builtin_amdgcn_cvt_pk_bf16_f32)
  auto pk = __builtin_amdgcn_cvt_pk_bf16_f32(lo, hi);  // v_cvt_pk_bf16_f32
  unsigned int u;
  __builtin_memcpy(&u, &pk, sizeof(unsigned int));
  return u;
#else
  return f2bf(lo) | (f2bf(hi) << 16);
#endif
#else
  return f2bf(lo) | (f2bf(hi) << 16);
#endif
}

// ---------------------------------------------------------------------------
// WMMA GEMM: C[M,N] = act(A[M,K] @ W[K,N] (+ bias))
// Requires K % 32 == 0 and N % 64 == 0 (true for every GEMM in this model).
// Block: 256 threads = 8 waves (4 M-waves x 2 N-waves), tile 128 x 64.
// Each wave owns a 32x32 sub-tile: 4 f32 accumulators, 4 WMMAs per K-step.
// A staged via float4 global_load_b128 -> packed bf16 ds_store_b64.
// ---------------------------------------------------------------------------
template<int RELU, int BIAS>
__global__ __launch_bounds__(256)
void gemm_wmma(const float* __restrict__ A, const float* __restrict__ W,
               const float* __restrict__ bias, float* __restrict__ C,
               int M, int N, int K) {
  __shared__ __align__(16) unsigned short Alds[128 * 32]; // 8 KB, row-major (K-minor)
  __shared__ __align__(16) unsigned short Blds[64 * 32];  // 4 KB, N-major (K-minor)

  const int t    = threadIdx.x;
  const int wave = t >> 5;
  const int lane = t & 31;
  const int h    = lane >> 4;   // half selector
  const int r    = lane & 15;
  const int mw   = wave >> 1;   // 0..3
  const int nw   = wave & 1;    // 0..1
  const int mBlock = blockIdx.x * 128;
  const int nBlock = blockIdx.y * 64;
  const bool fullM = (mBlock + 128 <= M);   // uniform per block

  // A-staging coordinates: 1024 float4s per tile, 4 per thread
  const int aRow0 = t >> 3;            // 0..31  (+32 per i)
  const int aCol4 = (t & 7) << 2;      // 0,4,...,28
  // B-staging coordinates: 512 float4s per tile, 2 per thread
  const int bK0  = t >> 4;             // 0..15 (+16 per i)
  const int bN4  = (t & 15) << 2;      // 0,4,...,60

  v8f acc00 = {0.f,0.f,0.f,0.f,0.f,0.f,0.f,0.f};
  v8f acc01 = {0.f,0.f,0.f,0.f,0.f,0.f,0.f,0.f};
  v8f acc10 = {0.f,0.f,0.f,0.f,0.f,0.f,0.f,0.f};
  v8f acc11 = {0.f,0.f,0.f,0.f,0.f,0.f,0.f,0.f};

  for (int k0 = 0; k0 < K; k0 += 32) {
    // ---- Stage A tile (128 rows x 32 k) as packed bf16 ----
    if (fullM) {
      #pragma unroll
      for (int i = 0; i < 4; ++i) {
        int row = aRow0 + i * 32;
        int gr  = mBlock + row;
        const float4 v = *(const float4*)&A[(size_t)gr * K + k0 + aCol4];
        uint2 p; p.x = pack2bf(v.x, v.y); p.y = pack2bf(v.z, v.w);
        *(uint2*)&Alds[row * 32 + aCol4] = p;
      }
    } else {
      #pragma unroll
      for (int i = 0; i < 4; ++i) {
        int row = aRow0 + i * 32;
        int gr  = mBlock + row;
        float4 v = make_float4(0.f, 0.f, 0.f, 0.f);
        if (gr < M) v = *(const float4*)&A[(size_t)gr * K + k0 + aCol4];
        uint2 p; p.x = pack2bf(v.x, v.y); p.y = pack2bf(v.z, v.w);
        *(uint2*)&Alds[row * 32 + aCol4] = p;
      }
    }
    // ---- Stage B tile (32 k x 64 n), transposed into N-major LDS ----
    #pragma unroll
    for (int i = 0; i < 2; ++i) {
      int kk = bK0 + i * 16;
      const float4 w = *(const float4*)&W[(size_t)(k0 + kk) * N + nBlock + bN4];
      Blds[(bN4 + 0) * 32 + kk] = (unsigned short)f2bf(w.x);
      Blds[(bN4 + 1) * 32 + kk] = (unsigned short)f2bf(w.y);
      Blds[(bN4 + 2) * 32 + kk] = (unsigned short)f2bf(w.z);
      Blds[(bN4 + 3) * 32 + kk] = (unsigned short)f2bf(w.w);
    }
    __syncthreads();

    // ---- Build fragments per ISA VGPR layouts (CDNA5 ISA) ----
    FragU a0, a1, b0, b1;
    const unsigned int* Au = (const unsigned int*)Alds; // 16 uints per row
    const unsigned int* Bu = (const unsigned int*)Blds; // 16 uints per col
    const int rowA0 = mw * 32 + r;
    const int rowA1 = rowA0 + 16;
    #pragma unroll
    for (int i = 0; i < 4; ++i) {
      a0.u[i]     = Au[rowA0 * 16 + h * 4 + i];      // K = h*8 + 0..7
      a0.u[4 + i] = Au[rowA0 * 16 + 8 + h * 4 + i];  // K = 16 + h*8 + 0..7
      a1.u[i]     = Au[rowA1 * 16 + h * 4 + i];
      a1.u[4 + i] = Au[rowA1 * 16 + 8 + h * 4 + i];
    }
    const int colB0 = nw * 32 + r;
    const int colB1 = colB0 + 16;
    #pragma unroll
    for (int i = 0; i < 8; ++i) {
      b0.u[i] = Bu[colB0 * 16 + (i >> 2) * 8 + h * 4 + (i & 3)];  // K = h*8 + 0..7, 16 + h*8 + 0..7
      b1.u[i] = Bu[colB1 * 16 + (i >> 2) * 8 + h * 4 + (i & 3)];
    }

    acc00 = __builtin_amdgcn_wmma_f32_16x16x32_bf16(false, a0.v, false, b0.v,
                                                    (short)0, acc00, false, false);
    acc01 = __builtin_amdgcn_wmma_f32_16x16x32_bf16(false, a0.v, false, b1.v,
                                                    (short)0, acc01, false, false);
    acc10 = __builtin_amdgcn_wmma_f32_16x16x32_bf16(false, a1.v, false, b0.v,
                                                    (short)0, acc10, false, false);
    acc11 = __builtin_amdgcn_wmma_f32_16x16x32_bf16(false, a1.v, false, b1.v,
                                                    (short)0, acc11, false, false);
    __syncthreads();
  }

  // Epilogue: C layout lane l -> N = l&15, M = (l>>4)*8 + v
  const int nOut0 = nBlock + nw * 32 + r;
  const int nOut1 = nOut0 + 16;
  const float bv0 = BIAS ? bias[nOut0] : 0.0f;
  const float bv1 = BIAS ? bias[nOut1] : 0.0f;
  #pragma unroll
  for (int v = 0; v < 8; ++v) {
    int gr0 = mBlock + mw * 32 + h * 8 + v;
    int gr1 = gr0 + 16;
    if (gr0 < M) {
      float v00 = acc00[v] + bv0, v01 = acc01[v] + bv1;
      if (RELU) { v00 = fmaxf(v00, 0.0f); v01 = fmaxf(v01, 0.0f); }
      C[(size_t)gr0 * N + nOut0] = v00;
      C[(size_t)gr0 * N + nOut1] = v01;
    }
    if (gr1 < M) {
      float v10 = acc10[v] + bv0, v11 = acc11[v] + bv1;
      if (RELU) { v10 = fmaxf(v10, 0.0f); v11 = fmaxf(v11, 0.0f); }
      C[(size_t)gr1 * N + nOut0] = v10;
      C[(size_t)gr1 * N + nOut1] = v11;
    }
  }
}

// ---------------------------------------------------------------------------
// Fill / elementwise kernels
// ---------------------------------------------------------------------------
__global__ void fill_f32(float* __restrict__ p, float v, size_t n) {
  for (size_t i = (size_t)blockIdx.x * blockDim.x + threadIdx.x; i < n;
       i += (size_t)gridDim.x * blockDim.x) p[i] = v;
}
__global__ void fill_u32(unsigned int* __restrict__ p, unsigned int v, size_t n) {
  for (size_t i = (size_t)blockIdx.x * blockDim.x + threadIdx.x; i < n;
       i += (size_t)gridDim.x * blockDim.x) p[i] = v;
}
__global__ void relu_inplace(float* __restrict__ p, size_t n) {
  for (size_t i = (size_t)blockIdx.x * blockDim.x + threadIdx.x; i < n;
       i += (size_t)gridDim.x * blockDim.x) p[i] = fmaxf(p[i], 0.0f);
}
// t1 = x1_res + concat(pe,pe,pe)
__global__ void add_pe(const float* __restrict__ xr, const float* __restrict__ pe,
                       float* __restrict__ out, int Nn) {
  size_t total = (size_t)Nn * 192;
  for (size_t i = (size_t)blockIdx.x * blockDim.x + threadIdx.x; i < total;
       i += (size_t)gridDim.x * blockDim.x) {
    size_t node = i / 192; int c = (int)(i % 192);
    out[i] = xr[i] + pe[node * 64 + (c & 63)];
  }
}
__global__ void add3(const float* __restrict__ a, const float* __restrict__ b,
                     const float* __restrict__ c, float* __restrict__ out, size_t n) {
  for (size_t i = (size_t)blockIdx.x * blockDim.x + threadIdx.x; i < n;
       i += (size_t)gridDim.x * blockDim.x) out[i] = a[i] + b[i] + c[i];
}

// ---------------------------------------------------------------------------
// Edge kernels (segment sums via global fp32 atomics)
// ---------------------------------------------------------------------------
__global__ void mpnn_scatter(const float* __restrict__ h, const int* __restrict__ src,
                             const int* __restrict__ dst, float* __restrict__ out,
                             int E, int C) {
  const int groups = C >> 2;
  const size_t total = (size_t)E * groups;
  for (size_t tid = (size_t)blockIdx.x * blockDim.x + threadIdx.x; tid < total;
       tid += (size_t)gridDim.x * blockDim.x) {
    int e = (int)(tid / groups);
    int c = (int)(tid % groups) << 2;
    int s = src[e], d = dst[e];
    const float4 v = *(const float4*)&h[(size_t)s * C + c];
    float* o = &out[(size_t)d * C + c];
    atomicAdd(o + 0, v.x); atomicAdd(o + 1, v.y);
    atomicAdd(o + 2, v.z); atomicAdd(o + 3, v.w);
  }
}
__global__ void gcn_scatter(const float* __restrict__ h, const int* __restrict__ src,
                            const int* __restrict__ dst, const float* __restrict__ dinv,
                            float* __restrict__ out, int E, int C) {
  const int groups = C >> 2;
  const size_t total = (size_t)E * groups;
  for (size_t tid = (size_t)blockIdx.x * blockDim.x + threadIdx.x; tid < total;
       tid += (size_t)gridDim.x * blockDim.x) {
    int e = (int)(tid / groups);
    int c = (int)(tid % groups) << 2;
    int s = src[e], d = dst[e];
    float norm = dinv[s] * dinv[d];
    const float4 v = *(const float4*)&h[(size_t)s * C + c];
    float* o = &out[(size_t)d * C + c];
    atomicAdd(o + 0, norm * v.x); atomicAdd(o + 1, norm * v.y);
    atomicAdd(o + 2, norm * v.z); atomicAdd(o + 3, norm * v.w);
  }
}
__global__ void deg_scatter(const int* __restrict__ dst, float* __restrict__ deg, int E) {
  for (size_t i = (size_t)blockIdx.x * blockDim.x + threadIdx.x; i < (size_t)E;
       i += (size_t)gridDim.x * blockDim.x) atomicAdd(&deg[dst[i]], 1.0f);
}
__global__ void make_dinv(const float* __restrict__ deg, float* __restrict__ dinv, int Nn) {
  for (int i = blockIdx.x * blockDim.x + threadIdx.x; i < Nn;
       i += gridDim.x * blockDim.x) {
    float d = deg[i] + 1.0f; // +1 for self-loop
    dinv[i] = (d > 0.0f) ? rsqrtf(d) : 0.0f;
  }
}
// out = relu(agg + dinv^2 * hg + bias)   (GCN self-loop + bias epilogue)
__global__ void gcn_finish(const float* __restrict__ agg, const float* __restrict__ hg,
                           const float* __restrict__ dinv, const float* __restrict__ bias,
                           float* __restrict__ out, int Nn, int C) {
  size_t total = (size_t)Nn * C;
  for (size_t i = (size_t)blockIdx.x * blockDim.x + threadIdx.x; i < total;
       i += (size_t)gridDim.x * blockDim.x) {
    size_t node = i / C; int c = (int)(i % C);
    float di = dinv[node];
    out[i] = fmaxf(agg[i] + di * di * hg[i] + bias[c], 0.0f);
  }
}

// ---------------------------------------------------------------------------
// Attentional aggregation (per-graph, per-channel softmax over nodes)
// ---------------------------------------------------------------------------
__device__ __forceinline__ void atomicMaxF(float* addr, float val) {
  if (val >= 0.0f) atomicMax((int*)addr, __float_as_int(val));
  else             atomicMin((unsigned int*)addr, __float_as_uint(val));
}
__global__ void att_max(const float* __restrict__ gate, const int* __restrict__ batch,
                        float* __restrict__ m, int Nn, int C) {
  size_t total = (size_t)Nn * C;
  for (size_t i = (size_t)blockIdx.x * blockDim.x + threadIdx.x; i < total;
       i += (size_t)gridDim.x * blockDim.x) {
    size_t node = i / C; int c = (int)(i % C);
    atomicMaxF(&m[(size_t)batch[node] * C + c], gate[i]);
  }
}
__global__ void att_expsum(const float* __restrict__ gate, const float* __restrict__ m,
                           const int* __restrict__ batch, float* __restrict__ ebuf,
                           float* __restrict__ ssum, int Nn, int C) {
  size_t total = (size_t)Nn * C;
  for (size_t i = (size_t)blockIdx.x * blockDim.x + threadIdx.x; i < total;
       i += (size_t)gridDim.x * blockDim.x) {
    size_t node = i / C; int c = (int)(i % C);
    int b = batch[node];
    float e = __expf(gate[i] - m[(size_t)b * C + c]);
    ebuf[i] = e;
    atomicAdd(&ssum[(size_t)b * C + c], e);
  }
}
__global__ void att_weight(const float* __restrict__ ebuf, const float* __restrict__ ssum,
                           const int* __restrict__ batch, const float* __restrict__ z,
                           float* __restrict__ zg, int Nn, int C) {
  size_t total = (size_t)Nn * C;
  for (size_t i = (size_t)blockIdx.x * blockDim.x + threadIdx.x; i < total;
       i += (size_t)gridDim.x * blockDim.x) {
    size_t node = i / C; int c = (int)(i % C);
    int b = batch[node];
    float a = ebuf[i] / ssum[(size_t)b * C + c];
    atomicAdd(&zg[(size_t)b * C + c], a * z[i]);
  }
}
// logits = relu(zg) @ Wo + bo ; softmax  (512 graphs x (192x10): scalar is fine)
__global__ void head_kernel(const float* __restrict__ zg, const float* __restrict__ Wo,
                            const float* __restrict__ bo, float* __restrict__ out, int G) {
  int g = blockIdx.x * blockDim.x + threadIdx.x;
  if (g >= G) return;
  float logits[10];
  #pragma unroll
  for (int j = 0; j < 10; ++j) logits[j] = bo[j];
  for (int k = 0; k < 192; ++k) {
    float zr = fmaxf(zg[(size_t)g * 192 + k], 0.0f);
    #pragma unroll
    for (int j = 0; j < 10; ++j) logits[j] += zr * Wo[k * 10 + j];
  }
  float mx = logits[0];
  #pragma unroll
  for (int j = 1; j < 10; ++j) mx = fmaxf(mx, logits[j]);
  float s = 0.0f;
  #pragma unroll
  for (int j = 0; j < 10; ++j) { logits[j] = __expf(logits[j] - mx); s += logits[j]; }
  float inv = 1.0f / s;
  #pragma unroll
  for (int j = 0; j < 10; ++j) out[(size_t)g * 10 + j] = logits[j] * inv;
}

// ---------------------------------------------------------------------------
// Host-side helpers
// ---------------------------------------------------------------------------
static inline int grid1d(size_t total, int block = 256, int cap = 1 << 18) {
  size_t b = (total + block - 1) / block;
  if (b > (size_t)cap) b = cap;
  if (b < 1) b = 1;
  return (int)b;
}
static inline void launch_gemm(int relu, int bias, const float* A, const float* W,
                               const float* b, float* C, int M, int N, int K,
                               hipStream_t s) {
  dim3 grid((M + 127) / 128, N / 64), block(256);
  if (relu) {
    if (bias) gemm_wmma<1,1><<<grid, block, 0, s>>>(A, W, b, C, M, N, K);
    else      gemm_wmma<1,0><<<grid, block, 0, s>>>(A, W, b, C, M, N, K);
  } else {
    if (bias) gemm_wmma<0,1><<<grid, block, 0, s>>>(A, W, b, C, M, N, K);
    else      gemm_wmma<0,0><<<grid, block, 0, s>>>(A, W, b, C, M, N, K);
  }
}

extern "C" void kernel_launch(void* const* d_in, const int* in_sizes, int n_in,
                              void* d_out, int out_size, void* d_ws, size_t ws_size,
                              hipStream_t stream) {
  // Inputs (setup_inputs order)
  const float* x1  = (const float*)d_in[0];
  const float* x2  = (const float*)d_in[1];
  const float* pe  = (const float*)d_in[2];
  const int*   ei  = (const int*)  d_in[3];
  const int*   bat = (const int*)  d_in[4];
  const float* Wc  = (const float*)d_in[5];  const float* bc  = (const float*)d_in[6];
  const float* Wm1 = (const float*)d_in[7];  const float* bm1 = (const float*)d_in[8];
  const float* Wg1 = (const float*)d_in[9];  const float* bg1 = (const float*)d_in[10];
  const float* Wm2 = (const float*)d_in[11]; const float* bm2 = (const float*)d_in[12];
  const float* Wg2 = (const float*)d_in[13]; const float* bg2 = (const float*)d_in[14];
  const float* Wa1 = (const float*)d_in[15]; const float* ba1 = (const float*)d_in[16];
  const float* Wa2 = (const float*)d_in[17]; const float* ba2 = (const float*)d_in[18];
  const float* Wo  = (const float*)d_in[19]; const float* bo  = (const float*)d_in[20];
  float* out = (float*)d_out;

  const int Nn = in_sizes[0] / 768;   // 100000 nodes
  const int E  = in_sizes[3] / 2;     // 1600000 edges
  const int G  = 512;
  const int* src = ei;
  const int* dst = ei + E;

  // Workspace carving (f32), lifetime-based aliasing
  float* r0   = (float*)d_ws;          // [N,192] x1_res            (A..G)
  float* r1   = r0 + (size_t)Nn * 192; // [N,256] t1 / deg / hg / hg2 / g1 / ebuf
  float* r2   = r1 + (size_t)Nn * 256; // [N,256] hm / hm2(+aggc)
  float* r3   = r2 + (size_t)Nn * 256; // [N,256] agg1->h1 / z
  float* r4   = r3 + (size_t)Nn * 256; // [N,192] gcn1 out
  float* r5   = r4 + (size_t)Nn * 192; // [N,192] c2 / gate
  float* dinv = r5 + (size_t)Nn * 192; // [N]
  float* m    = dinv + Nn;             // [G,192]
  float* ssum = m + (size_t)G * 192;   // [G,192]
  float* zg   = ssum + (size_t)G * 192;// [G,192]

  const int BL = 256;

  // ---- feature branch ----
  // x1_res = relu(x1 @ Wc + bc)
  launch_gemm(1, 1, x1, Wc, bc, r0, Nn, 192, 768, stream);
  // t1 = x1_res + concat(pe,pe,pe)
  add_pe<<<grid1d((size_t)Nn * 192), BL, 0, stream>>>(r0, pe, r1, Nn);
  // hm = t1 @ Wm1 + bm1   (bias inside message, as in reference)
  launch_gemm(0, 1, r1, Wm1, bm1, r2, Nn, 256, 192, stream);
  // h1 = relu(scatter_add(hm[src] -> dst))
  fill_f32<<<grid1d((size_t)Nn * 256), BL, 0, stream>>>(r3, 0.0f, (size_t)Nn * 256);
  mpnn_scatter<<<grid1d((size_t)E * 64), BL, 0, stream>>>(r2, src, dst, r3, E, 256);
  relu_inplace<<<grid1d((size_t)Nn * 256), BL, 0, stream>>>(r3, (size_t)Nn * 256);

  // degrees (shared by both GCNs): deg = indeg + 1 (self loop), dinv = deg^-1/2
  fill_f32<<<grid1d((size_t)Nn), BL, 0, stream>>>(r1, 0.0f, (size_t)Nn);
  deg_scatter<<<grid1d((size_t)E), BL, 0, stream>>>(dst, r1, E);
  make_dinv<<<grid1d((size_t)Nn), BL, 0, stream>>>(r1, dinv, Nn);

  // GCN1: hg = h1 @ Wg1; agg = norm-scatter; out = relu(agg + dinv^2*hg + bg1)
  launch_gemm(0, 0, r3, Wg1, nullptr, r1, Nn, 192, 256, stream);
  fill_f32<<<grid1d((size_t)Nn * 192), BL, 0, stream>>>(r4, 0.0f, (size_t)Nn * 192);
  gcn_scatter<<<grid1d((size_t)E * 48), BL, 0, stream>>>(r1, src, dst, dinv, r4, E, 192);
  gcn_finish<<<grid1d((size_t)Nn * 192), BL, 0, stream>>>(r4, r1, dinv, bg1, r4, Nn, 192);

  // ---- class branch ----
  // hm2 = x2 @ Wm2 + bm2 ; c1 = relu(scatter_add)
  launch_gemm(0, 1, x2, Wm2, bm2, r2, Nn, 64, 32, stream);
  float* c1 = r2 + (size_t)Nn * 64;
  fill_f32<<<grid1d((size_t)Nn * 64), BL, 0, stream>>>(c1, 0.0f, (size_t)Nn * 64);
  mpnn_scatter<<<grid1d((size_t)E * 16), BL, 0, stream>>>(r2, src, dst, c1, E, 64);
  relu_inplace<<<grid1d((size_t)Nn * 64), BL, 0, stream>>>(c1, (size_t)Nn * 64);
  // GCN2
  launch_gemm(0, 0, c1, Wg2, nullptr, r1, Nn, 192, 64, stream);
  fill_f32<<<grid1d((size_t)Nn * 192), BL, 0, stream>>>(r5, 0.0f, (size_t)Nn * 192);
  gcn_scatter<<<grid1d((size_t)E * 48), BL, 0, stream>>>(r1, src, dst, dinv, r5, E, 192);
  gcn_finish<<<grid1d((size_t)Nn * 192), BL, 0, stream>>>(r5, r1, dinv, bg2, r5, Nn, 192);

  // ---- fuse + attention ----
  // z = gcn1 + gcn2 + x1_res
  add3<<<grid1d((size_t)Nn * 192), BL, 0, stream>>>(r4, r5, r0, r3, (size_t)Nn * 192);
  // gate = relu(z@Wa1+ba1) @ Wa2 + ba2
  launch_gemm(1, 1, r3, Wa1, ba1, r1, Nn, 192, 192, stream);
  launch_gemm(0, 1, r1, Wa2, ba2, r5, Nn, 192, 192, stream);

  // attention pooling
  fill_u32<<<grid1d((size_t)G * 192), BL, 0, stream>>>((unsigned int*)m, 0xFF800000u,
                                                       (size_t)G * 192); // -inf
  fill_f32<<<grid1d((size_t)G * 192), BL, 0, stream>>>(ssum, 0.0f, (size_t)G * 192);
  fill_f32<<<grid1d((size_t)G * 192), BL, 0, stream>>>(zg, 0.0f, (size_t)G * 192);
  att_max<<<grid1d((size_t)Nn * 192), BL, 0, stream>>>(r5, bat, m, Nn, 192);
  att_expsum<<<grid1d((size_t)Nn * 192), BL, 0, stream>>>(r5, m, bat, r1, ssum, Nn, 192);
  att_weight<<<grid1d((size_t)Nn * 192), BL, 0, stream>>>(r1, ssum, bat, r3, zg, Nn, 192);

  // head: relu -> 192x10 linear -> softmax
  head_kernel<<<(G + 63) / 64, 64, 0, stream>>>(zg, Wo, bo, out, G);
}
